// MoELayer_67491116089917
// MI455X (gfx1250) — matrix operand order repaired and checked by reading, not compiled, not run on hardware
//
#include <hip/hip_runtime.h>
#include <hip/hip_bf16.h>
#include <math.h>

#define TOKENS 4096      // B*S = 2*2048
#define DIN    1024
#define DHID   4096
#define DOUT   1024
#define NEXP   8
#define TOPK   2

typedef __attribute__((ext_vector_type(16))) __bf16         v16bf;
typedef __attribute__((ext_vector_type(8)))  float          v8f;
typedef __attribute__((ext_vector_type(4)))  unsigned int   u32x4;
typedef __attribute__((ext_vector_type(4)))  unsigned short u16x4;
typedef __attribute__((ext_vector_type(4)))  float          f32x4;

union Frag { v16bf v; u32x4 u[2]; };
union Acc  { v8f v; float f[8]; };

__device__ __forceinline__ unsigned short f32_to_bf16(float f) {
    unsigned u = __float_as_uint(f);
    u += 0x7FFFu + ((u >> 16) & 1u);   // round-to-nearest-even
    return (unsigned short)(u >> 16);
}

// ---------------------------------------------------------------------------
// Kernel 1: zero expert counts + convert x (fp32) -> bf16 workspace copy
// ---------------------------------------------------------------------------
__global__ void prep_kernel(const float* __restrict__ x,
                            unsigned short* __restrict__ xbf,
                            int* __restrict__ counts) {
    int t = blockIdx.x * blockDim.x + threadIdx.x;
    if (t < NEXP) counts[t] = 0;
    const f32x4* x4 = (const f32x4*)x;
    u16x4* o4 = (u16x4*)xbf;
    const int n4 = TOKENS * DIN / 4;
    for (int i = t; i < n4; i += gridDim.x * blockDim.x) {
        f32x4 v = x4[i];
        u16x4 r;
        r.x = f32_to_bf16(v.x); r.y = f32_to_bf16(v.y);
        r.z = f32_to_bf16(v.z); r.w = f32_to_bf16(v.w);
        o4[i] = r;
    }
}

// ---------------------------------------------------------------------------
// Kernel 2: weight convert + transpose: W [E][K][N] f32 -> Wt [E][N][K] bf16
// 64x64 tiles through LDS so both global phases are coalesced.
// ---------------------------------------------------------------------------
__global__ __launch_bounds__(256)
void wconv_kernel(const float* __restrict__ W,
                  unsigned short* __restrict__ Wt,
                  int K, int N) {
    const int e = blockIdx.z;
    const int kBase = blockIdx.x * 64;
    const int nBase = blockIdx.y * 64;
    __shared__ unsigned short T[64 * 65];
    const float* Wp = W + (size_t)e * K * N;
    unsigned short* Wo = Wt + (size_t)e * N * K;
    const int t = threadIdx.x;
#pragma unroll
    for (int i = 0; i < 16; ++i) {
        int idx = t + i * 256;
        int kk = idx >> 6, nn = idx & 63;
        T[kk * 65 + nn] = f32_to_bf16(Wp[(size_t)(kBase + kk) * N + nBase + nn]);
    }
    __syncthreads();
#pragma unroll
    for (int i = 0; i < 4; ++i) {
        int q  = t + i * 256;     // 1024 quads of 4 halves
        int nn = q >> 4;
        int kq = (q & 15) * 4;
        u16x4 v;
        v.x = T[(kq + 0) * 65 + nn];
        v.y = T[(kq + 1) * 65 + nn];
        v.z = T[(kq + 2) * 65 + nn];
        v.w = T[(kq + 3) * 65 + nn];
        *(u16x4*)&Wo[(size_t)(nBase + nn) * K + kBase + kq] = v;
    }
}

// ---------------------------------------------------------------------------
// Kernel 3: gating — one wave32 per token
// ---------------------------------------------------------------------------
__global__ void gating_kernel(const float* __restrict__ x,
                              const float* __restrict__ Wg,
                              const float* __restrict__ bg,
                              float* __restrict__ logitsOut,   // [T][8]
                              int* __restrict__ idxOut,        // [T][2]
                              float* __restrict__ gateW,       // [T][2]
                              int* __restrict__ counts,        // [8]
                              int* __restrict__ pairTok) {     // [8][T] = token*2+k
    const int tok  = (blockIdx.x * blockDim.x + threadIdx.x) >> 5;
    const int lane = threadIdx.x & 31;
    if (tok >= TOKENS) return;

    const float* xr = x + (size_t)tok * DIN;
    float s[NEXP];
#pragma unroll
    for (int e = 0; e < NEXP; ++e) s[e] = 0.f;
    for (int i = lane; i < DIN; i += 32) {
        float xv = xr[i];
        const float* wr = Wg + (size_t)i * NEXP;
#pragma unroll
        for (int e = 0; e < NEXP; ++e) s[e] += xv * wr[e];
    }
#pragma unroll
    for (int e = 0; e < NEXP; ++e)
#pragma unroll
        for (int off = 16; off > 0; off >>= 1)
            s[e] += __shfl_xor(s[e], off, 32);

    if (lane == 0) {
        float l[NEXP];
#pragma unroll
        for (int e = 0; e < NEXP; ++e) {
            l[e] = s[e] + bg[e];
            logitsOut[tok * NEXP + e] = l[e];
        }
        int i0 = 0;
#pragma unroll
        for (int e = 1; e < NEXP; ++e) if (l[e] > l[i0]) i0 = e;
        int i1 = (i0 == 0) ? 1 : 0;
#pragma unroll
        for (int e = 0; e < NEXP; ++e) if (e != i0 && l[e] > l[i1]) i1 = e;
        float p1 = expf(l[i1] - l[i0]);
        float z  = 1.f + p1;
        idxOut[tok * 2 + 0] = i0;
        idxOut[tok * 2 + 1] = i1;
        gateW[tok * 2 + 0] = 1.f / z;
        gateW[tok * 2 + 1] = p1 / z;
        int s0 = atomicAdd(&counts[i0], 1);
        pairTok[i0 * TOKENS + s0] = tok * 2 + 0;
        int s1 = atomicAdd(&counts[i1], 1);
        pairTok[i1 * TOKENS + s1] = tok * 2 + 1;
    }
}

// ---------------------------------------------------------------------------
// Kernels 4/5: grouped expert GEMM, bf16 WMMA 16x16x32, fp32 accumulate.
// Block = 256 threads (8 waves). Tile: M=64 pairs x N=128 cols, K chunk 64.
//  - A (gathered token rows, bf16) staged in LDS, b128 copies, stride-72 pad.
//  - B fragments loaded DIRECTLY from pre-transposed bf16 weights Wt[E][N][K]:
//    each lane's fragment = two contiguous 16B global loads; wave w owns
//    N-columns [w*16, w*16+16) so there is no cross-wave duplication.
//  - 8 WMMAs per wave per K chunk (4 M-subtiles x 2 K-steps).
// ---------------------------------------------------------------------------
template <int KDIM, int NLD, bool GELU, bool ADIV2>
__global__ __launch_bounds__(256)
void moe_gemm_kernel(const unsigned short* __restrict__ Abase, // bf16, row stride KDIM
                     const unsigned short* __restrict__ Bt,    // [E][NLD][KDIM] bf16
                     const float* __restrict__ bias,           // [E][NLD]
                     const int* __restrict__ counts,
                     const int* __restrict__ pairTok,
                     unsigned short* __restrict__ outBf,       // gemm1: h [2T][DHID]
                     float* __restrict__ outF) {               // gemm2: [2T][NLD]
    const int e     = blockIdx.z;
    const int cnt   = counts[e];
    const int mBase = blockIdx.x * 64;
    if (mBase >= cnt) return;
    const int nBase = blockIdx.y * 128;
    const int tid   = threadIdx.x;
    const int lane  = tid & 31;
    const int w     = tid >> 5;

    __shared__ __align__(16) unsigned short Alds[64 * 72];
    __shared__ int rowEnc[64];

    if (tid < 64) {
        int slot = mBase + tid;
        rowEnc[tid] = (slot < cnt) ? pairTok[e * TOKENS + slot] : -1;
    }
    __syncthreads();

    Acc acc[4];
#pragma unroll
    for (int mi = 0; mi < 4; ++mi)
#pragma unroll
        for (int r = 0; r < 8; ++r) acc[mi].f[r] = 0.f;

    const int m   = lane & 15;
    const int hi8 = (lane & 16) ? 16 : 0;  // A K-segment select per lane half
    const int kb  = (lane & 16) ? 16 : 0;  // B K-base per lane half

    // A staging assignment: 4 threads per row, 16 halves (32B) each
    const int arow = tid >> 2;
    const int aseg = (tid & 3) * 16;
    const int aenc = rowEnc[arow];
    const unsigned short* aptr = nullptr;
    if (aenc >= 0)
        aptr = Abase + (size_t)(ADIV2 ? (aenc >> 1) : aenc) * KDIM + aseg;

    // This lane's B row (bf16, contiguous in K)
    const unsigned short* brow =
        Bt + ((size_t)e * NLD + nBase + w * 16 + m) * KDIM + kb;

    for (int kc = 0; kc < KDIM; kc += 64) {
        // ---- stage A tile: 64 x 64 bf16 (pure b128 copies) ----
        u32x4 a0 = {}, a1 = {};
        if (aptr) {
            a0 = *(const u32x4*)(aptr + kc);
            a1 = *(const u32x4*)(aptr + kc + 8);
        }
        *(u32x4*)&Alds[arow * 72 + aseg]     = a0;
        *(u32x4*)&Alds[arow * 72 + aseg + 8] = a1;
        if (kc + 64 < KDIM)
            __builtin_prefetch((const void*)(brow + kc + 64), 0, 1);
        __syncthreads();
#pragma unroll
        for (int ks = 0; ks < 2; ++ks) {
            Frag b;
            b.u[0] = *(const u32x4*)(brow + kc + ks * 32);
            b.u[1] = *(const u32x4*)(brow + kc + ks * 32 + 8);
#pragma unroll
            for (int mi = 0; mi < 4; ++mi) {
                Frag a;
                const int abase = (mi * 16 + m) * 72 + ks * 32 + hi8;
                a.u[0] = *(const u32x4*)&Alds[abase];
                a.u[1] = *(const u32x4*)&Alds[abase + 8];
                acc[mi].v = __builtin_amdgcn_wmma_f32_16x16x32_bf16(
                    false, a.v, false, b.v, (short)0, acc[mi].v, false, false);
            }
        }
        __syncthreads();
    }

    // ---- epilogue: bias (+ exact GELU -> bf16) or fp32 store ----
    const int n    = nBase + w * 16 + m;
    const int mOff = (lane & 16) ? 8 : 0;
    const float bval = bias[(size_t)e * NLD + n];
#pragma unroll
    for (int mi = 0; mi < 4; ++mi) {
#pragma unroll
        for (int r = 0; r < 8; ++r) {
            int enc = rowEnc[mi * 16 + r + mOff];
            if (enc < 0) continue;
            float v = acc[mi].f[r] + bval;
            if (GELU) {
                v = 0.5f * v * (1.0f + erff(v * 0.70710678118654752f));
                outBf[(size_t)enc * DHID + n] = f32_to_bf16(v);
            } else {
                outF[(size_t)enc * NLD + n] = v;
            }
        }
    }
}

// ---------------------------------------------------------------------------
// Kernel 6: combine the two expert outputs per token
// ---------------------------------------------------------------------------
__global__ void combine_kernel(const float* __restrict__ outPair, // [2T][DOUT]
                               const float* __restrict__ gateW,   // [T][2]
                               float* __restrict__ finalOut) {    // [T][DOUT]
    const int t = blockIdx.x;
    const float w0 = gateW[t * 2 + 0];
    const float w1 = gateW[t * 2 + 1];
    const f32x4* p0 = (const f32x4*)(outPair + (size_t)(t * 2 + 0) * DOUT);
    const f32x4* p1 = (const f32x4*)(outPair + (size_t)(t * 2 + 1) * DOUT);
    f32x4* o = (f32x4*)(finalOut + (size_t)t * DOUT);
    for (int c = threadIdx.x; c < DOUT / 4; c += blockDim.x)
        o[c] = w0 * p0[c] + w1 * p1[c];
}

// ---------------------------------------------------------------------------
extern "C" void kernel_launch(void* const* d_in, const int* in_sizes, int n_in,
                              void* d_out, int out_size, void* d_ws, size_t ws_size,
                              hipStream_t stream) {
    const float* x  = (const float*)d_in[0];
    const float* W1 = (const float*)d_in[1];
    const float* b1 = (const float*)d_in[2];
    const float* W2 = (const float*)d_in[3];
    const float* b2 = (const float*)d_in[4];
    const float* Wg = (const float*)d_in[5];
    const float* bg = (const float*)d_in[6];

    float* outFinal  = (float*)d_out;                               // [T][DOUT]
    float* outLogits = outFinal + (size_t)TOKENS * DOUT;            // [T][8]
    int*   outIdx    = (int*)(outLogits + (size_t)TOKENS * NEXP);   // [T][2]

    // workspace layout (256B-aligned regions)
    uint8_t* ws = (uint8_t*)d_ws;
    int*            counts  = (int*)(ws + 0);                   //     32 B
    int*            pairTok = (int*)(ws + 256);                 //  128 KB
    float*          gateW   = (float*)(ws + 131328);            //   32 KB
    unsigned short* xbf     = (unsigned short*)(ws + 164096);   //    8 MB
    unsigned short* hbuf    = (unsigned short*)(ws + 8552704);  //   64 MB [2T][DHID] bf16
    float*          outPair = (float*)(ws + 75661568);          //   32 MB [2T][DOUT] f32
    unsigned short* W1t     = (unsigned short*)(ws + 109216000);//   64 MB [E][DHID][DIN]
    unsigned short* W2t     = (unsigned short*)(ws + 176324864);//   64 MB [E][DOUT][DHID]

    prep_kernel<<<2048, 256, 0, stream>>>(x, xbf, counts);

    // one-pass weight convert+transpose to bf16 [E][N][K]
    wconv_kernel<<<dim3(DIN / 64, DHID / 64, NEXP), 256, 0, stream>>>(W1, W1t, DIN, DHID);
    wconv_kernel<<<dim3(DHID / 64, DOUT / 64, NEXP), 256, 0, stream>>>(W2, W2t, DHID, DOUT);

    gating_kernel<<<(TOKENS * 32) / 256, 256, 0, stream>>>(
        x, Wg, bg, outLogits, outIdx, gateW, counts, pairTok);

    // GEMM1: h = gelu(x @ W1 + b1), per routed (token, expert) pair
    moe_gemm_kernel<DIN, DHID, true, true>
        <<<dim3(TOKENS / 64, DHID / 128, NEXP), 256, 0, stream>>>(
            xbf, W1t, b1, counts, pairTok, hbuf, nullptr);

    // GEMM2: pair_out = h @ W2 + b2
    moe_gemm_kernel<DHID, DOUT, false, false>
        <<<dim3(TOKENS / 64, DOUT / 128, NEXP), 256, 0, stream>>>(
            hbuf, W2t, b2, counts, pairTok, nullptr, outPair);

    combine_kernel<<<TOKENS, 256, 0, stream>>>(outPair, gateW, outFinal);
}
